// Attn_15968688406744
// MI455X (gfx1250) — matrix operand order repaired and checked by reading, not compiled, not run on hardware
//
#include <hip/hip_runtime.h>

typedef __attribute__((ext_vector_type(16))) __bf16 v16bf;
typedef __attribute__((ext_vector_type(2)))  __bf16 v2bf;
typedef __attribute__((ext_vector_type(8)))  float  v8f;

struct alignas(16) U4 { unsigned int x, y, z, w; };
union FragU { v16bf v; U4 q4[2]; unsigned int w[8]; };
template <bool M> struct BC { static constexpr bool value = M; };

constexpr int B  = 2, H = 32, HK = 8, G = 4, NQ = 2048, SQ = 2048, D = 128;
constexpr int BM = 256;         // query rows per workgroup (32 per wave)
constexpr int BN = 64;          // keys per softmax cadence (4 x 16-key tiles)
constexpr int NW = 8;           // waves per workgroup (wave32 -> 256 threads)
constexpr int KSTR = D + 8;     // LDS row strides (ushorts), padded vs bank conflicts
constexpr int VSTR = BN + 8;
constexpr int PSTR = BN + 8;

// hardware bf16 conversion (clang lowers f32->bf16 natively on gfx1250)
__device__ __forceinline__ unsigned short f2bf(float f) {
  return __builtin_bit_cast(unsigned short, (__bf16)f);
}
__device__ __forceinline__ unsigned int pk2(float a, float b) {
  v2bf t = {(__bf16)a, (__bf16)b};
  return __builtin_bit_cast(unsigned int, t);
}
// compare-select max: avoids fmaxf's canonicalize self-max ops (scores are finite)
__device__ __forceinline__ float maxsel(float a, float b) { return a > b ? a : b; }
// single-instruction lane-XOR shuffle: ds_swizzle_b32, group-of-32 xor mode
template <int M>
__device__ __forceinline__ float swz(float x) {
  int i = __builtin_amdgcn_ds_swizzle(__builtin_bit_cast(int, x), (M << 10) | 0x1f);
  return __builtin_bit_cast(float, i);
}

__global__ __launch_bounds__(NW * 32)
void fa_fwd(const float* __restrict__ q, const float* __restrict__ k,
            const float* __restrict__ v, float* __restrict__ o) {
  // single-buffered 64-key K (row-major) / V (transposed) tiles in bf16
  __shared__ alignas(16) unsigned short Ksh[BN][KSTR];
  __shared__ alignas(16) unsigned short VshT[D][VSTR];
  __shared__ alignas(16) unsigned short Psh[NW][16][PSTR];  // one M-tile of P per wave

  const int bh = blockIdx.x;           // [0, B*H)
  const int b  = bh / H;
  const int hq = bh % H;
  const int hk = hq / G;               // '(h g)' grouping: consecutive G heads share a kv head
  const int q0 = blockIdx.y * BM;

  // wave id as a proven-uniform scalar -> causal skip becomes an SGPR branch
  const int wq   = __builtin_amdgcn_readfirstlane(threadIdx.x >> 5);
  const int lane = threadIdx.x & 31;
  const int ln   = lane & 15;
  const int hi   = lane >> 4;          // 0 or 1 (half-wave)
  const int wrow = q0 + wq * 32;       // this wave's first query row (scalar)

  const float scale = 0.088388347648318447f * 1.4426950408889634f; // 1/sqrt(D) * log2(e)

  // ---- Q fragments: two 16-row M-tiles, 16-bit A-matrix layout, scale folded in ----
  FragU qf[2][4];
  #pragma unroll
  for (int mt = 0; mt < 2; ++mt) {
    const float* qrow = q + ((size_t)bh * NQ + (wrow + mt * 16 + ln)) * D;
    #pragma unroll
    for (int kk = 0; kk < 4; ++kk) {
      const int b1 = kk * 32 + 8 * hi;
      float4 a0 = *(const float4*)(qrow + b1);
      float4 a1 = *(const float4*)(qrow + b1 + 4);
      float4 c0 = *(const float4*)(qrow + b1 + 16);
      float4 c1 = *(const float4*)(qrow + b1 + 20);
      qf[mt][kk].w[0] = pk2(a0.x * scale, a0.y * scale);
      qf[mt][kk].w[1] = pk2(a0.z * scale, a0.w * scale);
      qf[mt][kk].w[2] = pk2(a1.x * scale, a1.y * scale);
      qf[mt][kk].w[3] = pk2(a1.z * scale, a1.w * scale);
      qf[mt][kk].w[4] = pk2(c0.x * scale, c0.y * scale);
      qf[mt][kk].w[5] = pk2(c0.z * scale, c0.w * scale);
      qf[mt][kk].w[6] = pk2(c1.x * scale, c1.y * scale);
      qf[mt][kk].w[7] = pk2(c1.z * scale, c1.w * scale);
    }
  }

  // all-ones B fragment: row-sum of P via the matrix pipe (l = P x ones)
  FragU ones;
  #pragma unroll
  for (int i = 0; i < 8; ++i) ones.w[i] = 0x3F803F80u;

  const v8f vzero = {0.f, 0.f, 0.f, 0.f, 0.f, 0.f, 0.f, 0.f};
  v8f acc[2][8];   // O accumulators
  v8f acc_l[2];    // row-sum accumulators (softmax denominator), same alpha path as O
  float m_i[2][8];
  #pragma unroll
  for (int mt = 0; mt < 2; ++mt) {
    #pragma unroll
    for (int dt = 0; dt < 8; ++dt) acc[mt][dt] = vzero;
    acc_l[mt] = vzero;
    #pragma unroll
    for (int r = 0; r < 8; ++r) m_i[mt][r] = -1e30f;
  }

  const float* kbase = k + (size_t)(b * HK + hk) * SQ * D;
  const float* vbase = v + (size_t)(b * HK + hk) * SQ * D;

  const int nkb = (q0 + BM) / BN;          // causal: keys [0, q0+BM)
  const int tr  = threadIdx.x >> 2;        // staging row 0..63
  const int tc  = (threadIdx.x & 3) * 32;  // staging col base (32 floats per thread)

  float4 kp[8], vp[8];
  auto issue = [&](int s0n) {              // global loads of a key tile -> registers
    const float* kr = kbase + (size_t)(s0n + tr) * D + tc;
    const float* vr = vbase + (size_t)(s0n + tr) * D + tc;
    #pragma unroll
    for (int j = 0; j < 8; ++j) {
      kp[j] = *(const float4*)(kr + 4 * j);
      vp[j] = *(const float4*)(vr + 4 * j);
    }
  };
  auto commit = [&]() {                    // convert + store registers -> LDS
    #pragma unroll
    for (int j = 0; j < 8; ++j) {
      *(unsigned int*)&Ksh[tr][tc + 4 * j]     = pk2(kp[j].x, kp[j].y);
      *(unsigned int*)&Ksh[tr][tc + 4 * j + 2] = pk2(kp[j].z, kp[j].w);
      VshT[tc + 4 * j    ][tr] = f2bf(vp[j].x);
      VshT[tc + 4 * j + 1][tr] = f2bf(vp[j].y);
      VshT[tc + 4 * j + 2][tr] = f2bf(vp[j].z);
      VshT[tc + 4 * j + 3][tr] = f2bf(vp[j].w);
    }
  };

  issue(0);
  commit();
  __syncthreads();

  for (int kb = 0; kb < nkb; ++kb) {
    const int s0  = kb * BN;
    const bool more = (kb + 1) < nkb;

    // next tile's global loads stay in flight during compute
    if (more) issue(s0 + BN);
    if (kb + 2 < nkb) {  // L2 warm-up two tiles ahead (global_prefetch_b8)
      __builtin_prefetch(kbase + (size_t)(s0 + 2 * BN + tr) * D + tc, 0, 1);
      __builtin_prefetch(vbase + (size_t)(s0 + 2 * BN + tr) * D + tc, 0, 1);
    }

    // per-wave causal skip (scalar condition -> s_cbranch; EXEC stays all-ones inside)
    if (s0 < wrow + 32) {
      // ---- S = Q K^T : 2 M-tiles x 4 key-tiles; each K-frag feeds 2 WMMAs ----
      v8f sc[2][4];
      #pragma unroll
      for (int mt = 0; mt < 2; ++mt)
        #pragma unroll
        for (int kt = 0; kt < 4; ++kt) sc[mt][kt] = vzero;
      #pragma unroll
      for (int kk = 0; kk < 4; ++kk) {
        #pragma unroll
        for (int kt = 0; kt < 4; ++kt) {
          FragU kf;
          kf.q4[0] = *(const U4*)&Ksh[kt * 16 + ln][kk * 32 + 8 * hi];
          kf.q4[1] = *(const U4*)&Ksh[kt * 16 + ln][kk * 32 + 16 + 8 * hi];
          sc[0][kt] = __builtin_amdgcn_wmma_f32_16x16x32_bf16(false, qf[0][kk].v, false, kf.v,
                                                              (short)0, sc[0][kt], false, false);
          sc[1][kt] = __builtin_amdgcn_wmma_f32_16x16x32_bf16(false, qf[1][kk].v, false, kf.v,
                                                              (short)0, sc[1][kt], false, false);
        }
      }

      // ---- online softmax over 64 columns; sum deferred to matrix pipe ----
      FragU pf[2][2];  // per M-tile, two A-fragments (k=0..31, k=32..63)
      auto softstep = [&](int mt, auto maskc) {
        constexpr bool MASK = decltype(maskc)::value;
        const int mrow0 = wrow + mt * 16 + 8 * hi;
        v8f alv;
        #pragma unroll
        for (int r = 0; r < 8; ++r) {
          const int mg = mrow0 + r;
          float x[4];
          #pragma unroll
          for (int kt = 0; kt < 4; ++kt) {
            x[kt] = sc[mt][kt][r];
            if (MASK) x[kt] = ((s0 + kt * 16 + ln) <= mg) ? x[kt] : -1e30f;
          }
          float t = maxsel(maxsel(x[0], x[1]), maxsel(x[2], x[3]));
          t = maxsel(t, swz<1>(t));
          t = maxsel(t, swz<2>(t));
          t = maxsel(t, swz<4>(t));
          t = maxsel(t, swz<8>(t));
          const float mnew = maxsel(m_i[mt][r], t);
          alv[r] = __builtin_amdgcn_exp2f(m_i[mt][r] - mnew);
          m_i[mt][r] = mnew;
          #pragma unroll
          for (int kt = 0; kt < 4; ++kt)
            Psh[wq][r + 8 * hi][kt * 16 + ln] = f2bf(__builtin_amdgcn_exp2f(x[kt] - mnew));
        }
        // vector alpha-rescale (v_pk_mul_f32-friendly)
        acc_l[mt] *= alv;
        #pragma unroll
        for (int dt = 0; dt < 8; ++dt) acc[mt][dt] *= alv;
        // P fragments for this M-tile (before Psh is reused by the next one)
        pf[mt][0].q4[0] = *(const U4*)&Psh[wq][ln][16 * hi];
        pf[mt][0].q4[1] = *(const U4*)&Psh[wq][ln][16 * hi + 8];
        pf[mt][1].q4[0] = *(const U4*)&Psh[wq][ln][32 + 16 * hi];
        pf[mt][1].q4[1] = *(const U4*)&Psh[wq][ln][32 + 16 * hi + 8];
      };
      if (s0 + BN - 1 > wrow) { softstep(0, BC<true>{});  softstep(1, BC<true>{});  }
      else                    { softstep(0, BC<false>{}); softstep(1, BC<false>{}); }

      // ---- l += P x ones ; O += P V : each V-frag feeds 2 WMMAs ----
      acc_l[0] = __builtin_amdgcn_wmma_f32_16x16x32_bf16(false, pf[0][0].v, false, ones.v,
                                                         (short)0, acc_l[0], false, false);
      acc_l[0] = __builtin_amdgcn_wmma_f32_16x16x32_bf16(false, pf[0][1].v, false, ones.v,
                                                         (short)0, acc_l[0], false, false);
      acc_l[1] = __builtin_amdgcn_wmma_f32_16x16x32_bf16(false, pf[1][0].v, false, ones.v,
                                                         (short)0, acc_l[1], false, false);
      acc_l[1] = __builtin_amdgcn_wmma_f32_16x16x32_bf16(false, pf[1][1].v, false, ones.v,
                                                         (short)0, acc_l[1], false, false);
      #pragma unroll
      for (int dt = 0; dt < 8; ++dt) {
        FragU vf0, vf1;
        const U4* p0 = (const U4*)&VshT[dt * 16 + ln][16 * hi];
        const U4* p1 = (const U4*)&VshT[dt * 16 + ln][32 + 16 * hi];
        vf0.q4[0] = p0[0]; vf0.q4[1] = p0[1];
        vf1.q4[0] = p1[0]; vf1.q4[1] = p1[1];
        acc[0][dt] = __builtin_amdgcn_wmma_f32_16x16x32_bf16(false, pf[0][0].v, false, vf0.v,
                                                             (short)0, acc[0][dt], false, false);
        acc[0][dt] = __builtin_amdgcn_wmma_f32_16x16x32_bf16(false, pf[0][1].v, false, vf1.v,
                                                             (short)0, acc[0][dt], false, false);
        acc[1][dt] = __builtin_amdgcn_wmma_f32_16x16x32_bf16(false, pf[1][0].v, false, vf0.v,
                                                             (short)0, acc[1][dt], false, false);
        acc[1][dt] = __builtin_amdgcn_wmma_f32_16x16x32_bf16(false, pf[1][1].v, false, vf1.v,
                                                             (short)0, acc[1][dt], false, false);
      }
    }

    __syncthreads();          // all waves done reading Ksh/VshT
    if (more) commit();       // overwrite the single buffer
    __syncthreads();          // commit visible to all waves
  }

  // ---- normalize and store (fp32); acc_l holds the row sum replicated per lane ----
  #pragma unroll
  for (int mt = 0; mt < 2; ++mt) {
    float* orow0 = o + ((size_t)bh * NQ + (wrow + mt * 16 + 8 * hi)) * D;
    #pragma unroll
    for (int r = 0; r < 8; ++r) {
      const float inv = __builtin_amdgcn_rcpf(acc_l[mt][r]);
      float* orow = orow0 + (size_t)r * D;
      #pragma unroll
      for (int dt = 0; dt < 8; ++dt)
        orow[dt * 16 + ln] = acc[mt][dt][r] * inv;
    }
  }
}

extern "C" void kernel_launch(void* const* d_in, const int* in_sizes, int n_in,
                              void* d_out, int out_size, void* d_ws, size_t ws_size,
                              hipStream_t stream) {
  (void)in_sizes; (void)n_in; (void)out_size; (void)d_ws; (void)ws_size;
  const float* q = (const float*)d_in[0];
  const float* k = (const float*)d_in[1];
  const float* v = (const float*)d_in[2];
  float* o = (float*)d_out;
  dim3 grid(B * H, NQ / BM);
  fa_fwd<<<grid, NW * 32, 0, stream>>>(q, k, v, o);
}
